// TRM_MoR_Inner_28759101014513
// MI455X (gfx1250) — compile-verified, matrix operand order checked by reading, not run on hardware
//
#include <hip/hip_runtime.h>

typedef unsigned short u16;
typedef unsigned int   u32;
typedef __bf16 bf16;
typedef bf16 v16bf __attribute__((ext_vector_type(16)));
typedef float v8f   __attribute__((ext_vector_type(8)));
typedef float f32x4 __attribute__((ext_vector_type(4)));
typedef u32   u32x4 __attribute__((ext_vector_type(4)));
typedef int   v4i   __attribute__((ext_vector_type(4)));

#define B_     32
#define S_     2048
#define H_     512
#define V_     128
#define P_     256
#define M_     128
#define SLOTS_ 64
#define L_     2
#define IT_    2816
#define IM_    768
#define EPS_   1e-5f
#define SQRT_H 22.627416997969522f

#if defined(__has_builtin)
#  if __has_builtin(__builtin_amdgcn_global_load_async_to_lds_b128) && \
      __has_builtin(__builtin_amdgcn_s_wait_asynccnt)
#    define HAVE_ASYNC 1
#  else
#    define HAVE_ASYNC 0
#  endif
#else
#  define HAVE_ASYNC 0
#endif

#if HAVE_ASYNC
// 16B global -> LDS async copy (ASYNCcnt path). AS1/AS3 pointers built via
// integer casts: low 32 bits of a generic shared pointer == LDS byte offset.
__device__ __forceinline__ void async_cp16(const u16* g, u16* l) {
  __builtin_amdgcn_global_load_async_to_lds_b128(
      (__attribute__((address_space(1))) v4i*)(unsigned long long)g,
      (__attribute__((address_space(3))) v4i*)(unsigned long long)l,
      0, 0);
}
#endif

__device__ __forceinline__ u16 f2bf(float f) {
  unsigned u = __float_as_uint(f);
  u += 0x7FFFu + ((u >> 16) & 1u);   // round-to-nearest-even
  return (u16)(u >> 16);
}

// fp32 -> bf16 bulk converter (8 elements / thread, 16B stores)
__global__ __launch_bounds__(256) void cvt_k(const float* __restrict__ src,
                                             u16* __restrict__ dst) {
  long long i = (long long)blockIdx.x * blockDim.x + threadIdx.x;
  const f32x4* p = (const f32x4*)src;
  f32x4 a = p[2 * i], b = p[2 * i + 1];
  u32x4 r;
  r.x = (u32)f2bf(a.x) | ((u32)f2bf(a.y) << 16);
  r.y = (u32)f2bf(a.z) | ((u32)f2bf(a.w) << 16);
  r.z = (u32)f2bf(b.x) | ((u32)f2bf(b.y) << 16);
  r.w = (u32)f2bf(b.z) | ((u32)f2bf(b.w) << 16);
  ((u32x4*)dst)[i] = r;
}

// ---------------------------------------------------------------------------
// WMMA GEMM:  C[r,n] = epilogue( sum_k A[r,k] * W[n,k] ),  W is bf16 (N,K)
// Double-buffered LDS staging: async copy of chunk c+1 overlaps WMMA of c.
// ABF=false: A fp32, element addr = (r>>rshift)*sB + (r&mask)*sR + k*sK
// ABF=true : A bf16 row-major with leading dim ldA
// MODE 0: dual-set gate/up, out16 = bf16( silu(g)*u )   (W rows n and wOff2+n)
// MODE 1: outF = acc + resid[b, n, h]   (r = b*H+h, strided residual)
// MODE 2: outF = acc + resid[r*H + n]   (row-major residual)
// MODE 3: outF = acc                    (lm_head)
// ---------------------------------------------------------------------------
#define BLK_M 128
#define BLK_N 64
#define BLK_K 32
#define LDT   40   // LDS pitch (bf16 elems): 80B rows -> conflict-free b128 reads

union Frag { v16bf v; f32x4 q[2]; };

template <int MODE, bool ABF>
__global__ __launch_bounds__(256) void gemm_wmma_k(
    const float* __restrict__ Af, const u16* __restrict__ A16, int ldA,
    const u16* __restrict__ W16,
    float* __restrict__ outF, u16* __restrict__ out16,
    int rshift, long long sB, long long sR, long long sK,
    int K, const float* __restrict__ resid,
    long long wOff2, int outLd)
{
  constexpr int NSET = (MODE == 0) ? 2 : 1;
  constexpr int GASYNC = (ABF ? 2 : 0) + NSET;   // async loads issued per chunk
  __shared__ u16 As[2][BLK_M * LDT];
  __shared__ u16 Bs[2][2][BLK_N * LDT];

  const int t    = threadIdx.x;
  const int lane = t & 31;
  const int wid  = t >> 5;
  const int wm   = (wid & 3) * 32;   // wave row offset in C tile
  const int wn   = (wid >> 2) * 32;  // wave col offset in C tile
  const int n0   = blockIdx.x * BLK_N;
  const int m0   = blockIdx.y * BLK_M;

  // Precompute fp32-A source addresses (handles the transposed z_L view).
  long long abase[8];
  if constexpr (!ABF) {
    const long long mask = (1LL << rshift) - 1;
#pragma unroll
    for (int i = 0; i < 8; ++i) {
      int idx = t + i * 256;
      long long r = m0 + (idx >> 4);
      int kp = (idx & 15) * 2;
      abase[i] = ((r >> rshift) * sB) + ((r & mask) * sR) + (long long)kp * sK;
    }
  }

  // Stage one K-chunk (kc) into LDS buffer bufSel.
  auto stage = [&](int kc, int bufSel) {
    if constexpr (ABF) {
#pragma unroll
      for (int i = 0; i < 2; ++i) {
        int idx = t + i * 256;
        int row = idx >> 2, kq = (idx & 3) * 8;
        const u16* g = A16 + (long long)(m0 + row) * ldA + kc + kq;
        u16* l = &As[bufSel][row * LDT + kq];
#if HAVE_ASYNC
        async_cp16(g, l);
#else
        *reinterpret_cast<f32x4*>(l) = *reinterpret_cast<const f32x4*>(g);
#endif
      }
    } else {
#pragma unroll
      for (int i = 0; i < 8; ++i) {
        int idx = t + i * 256;
        int row = idx >> 4, kp = (idx & 15) * 2;
        long long base = abase[i] + (long long)kc * sK;
        u32 lo = f2bf(Af[base]);
        u32 hi = f2bf(Af[base + sK]);
        *reinterpret_cast<u32*>(&As[bufSel][row * LDT + kp]) = lo | (hi << 16);
      }
    }
#pragma unroll
    for (int s = 0; s < NSET; ++s) {
      int row = t >> 2, kq = (t & 3) * 8;
      const u16* g = W16 + ((long long)n0 + (s ? wOff2 : 0) + row) * K + kc + kq;
      u16* l = &Bs[bufSel][s][row * LDT + kq];
#if HAVE_ASYNC
      async_cp16(g, l);
#else
      *reinterpret_cast<f32x4*>(l) = *reinterpret_cast<const f32x4*>(g);
#endif
    }
  };

  v8f acc[NSET][2][2];
  v8f vzero = {0.f, 0.f, 0.f, 0.f, 0.f, 0.f, 0.f, 0.f};
#pragma unroll
  for (int s = 0; s < NSET; ++s)
#pragma unroll
    for (int mi = 0; mi < 2; ++mi)
#pragma unroll
      for (int ni = 0; ni < 2; ++ni) acc[s][mi][ni] = vzero;

  const int lm = lane & 15;
  const int kb = (lane >> 4) * 8;    // K base per lane half (ISA bf16 layout)
  const int nch = K / BLK_K;

  stage(0, 0);                       // prologue: chunk 0 -> buffer 0

  for (int c = 0; c < nch; ++c) {
    const int cur = c & 1;
    if (c + 1 < nch) {
      stage((c + 1) * BLK_K, cur ^ 1);   // overlap next chunk's copy
#if HAVE_ASYNC
      __builtin_amdgcn_s_wait_asynccnt(GASYNC);  // chunk c landed
#endif
    } else {
#if HAVE_ASYNC
      __builtin_amdgcn_s_wait_asynccnt(0);
#endif
    }
    __syncthreads();   // publish buf[cur] to all waves

    Frag a[2], b[NSET][2];
#pragma unroll
    for (int mi = 0; mi < 2; ++mi) {
      const f32x4* pa = reinterpret_cast<const f32x4*>(
          &As[cur][(wm + mi * 16 + lm) * LDT + kb]);
      a[mi].q[0] = pa[0];   // K = kb .. kb+7
      a[mi].q[1] = pa[2];   // K = kb+16 .. kb+23
    }
#pragma unroll
    for (int s = 0; s < NSET; ++s)
#pragma unroll
      for (int ni = 0; ni < 2; ++ni) {
        const f32x4* pb = reinterpret_cast<const f32x4*>(
            &Bs[cur][s][(wn + ni * 16 + lm) * LDT + kb]);
        b[s][ni].q[0] = pb[0];
        b[s][ni].q[1] = pb[2];
      }
#pragma unroll
    for (int s = 0; s < NSET; ++s)
#pragma unroll
      for (int mi = 0; mi < 2; ++mi)
#pragma unroll
        for (int ni = 0; ni < 2; ++ni)
          acc[s][mi][ni] = __builtin_amdgcn_wmma_f32_16x16x32_bf16(
              false, a[mi].v, false, b[s][ni].v, (short)0, acc[s][mi][ni],
              false, false);
    __syncthreads();   // ds loads drained (dscnt 0) before buffer reuse
  }

  // Epilogue (C layout: col = lane%16, row = vgpr + 8*(lane>=16))
#pragma unroll
  for (int mi = 0; mi < 2; ++mi)
#pragma unroll
    for (int ni = 0; ni < 2; ++ni)
#pragma unroll
      for (int rr = 0; rr < 8; ++rr) {
        int gm = m0 + wm + mi * 16 + ((lane >> 4) << 3) + rr;
        int gn = n0 + wn + ni * 16 + lm;
        if constexpr (MODE == 0) {
          float g = acc[0][mi][ni][rr];
          float u = acc[NSET - 1][mi][ni][rr];
          out16[(long long)gm * outLd + gn] = f2bf((g / (1.f + __expf(-g))) * u);
        } else if constexpr (MODE == 1) {
          int bb = gm >> 9, hh = gm & (H_ - 1);   // r = b*H + h
          outF[(long long)gm * outLd + gn] =
              acc[0][mi][ni][rr] + resid[((long long)bb * S_ + gn) * H_ + hh];
        } else if constexpr (MODE == 2) {
          outF[(long long)gm * outLd + gn] =
              acc[0][mi][ni][rr] + resid[(long long)gm * H_ + gn];
        } else {
          outF[(long long)gm * outLd + gn] = acc[0][mi][ni][rr];
        }
      }
}

// ---------------------------------------------------------------------------
// pooled[b,h] = mean_s z[b,s,h]
__global__ void pool_k(const float* __restrict__ z, float* __restrict__ pooled) {
  int g = blockIdx.x * blockDim.x + threadIdx.x;   // < B*H
  int b = g >> 9, hh = g & (H_ - 1);
  const float* p = z + (long long)b * S_ * H_ + hh;
  float s = 0.f;
  for (int ss = 0; ss < S_; ++ss) s += p[(long long)ss * H_];
  pooled[g] = s * (1.f / S_);
}

// Planner MLP + memory read/write (one block per batch element)
__global__ __launch_bounds__(256) void planner_k(
    const float* __restrict__ pooled, float* __restrict__ hbuf,
    float* __restrict__ memw, float* __restrict__ film_g,
    float* __restrict__ ctx_g, float* __restrict__ haltOut,
    const float* pw1, const float* pb1, const float* pw2, const float* pb2,
    const float* hdw, const float* hdb, const float* fw, const float* fb,
    const float* hw, const float* hb,
    const float* mqw, const float* mqb, const float* mkw, const float* mkb,
    const float* mvw, const float* mvb, const float* mgw, const float* mgb,
    const float* mow, const float* mob)
{
  const int b = blockIdx.x, t = threadIdx.x;
  __shared__ float cat[H_ + P_];
  __shared__ float t1s[P_];
  __shared__ float hm[P_];
  __shared__ float qv[M_], kv[M_], vv[M_];
  __shared__ float sc[SLOTS_], wsc[SLOTS_];
  __shared__ float ctxm[M_];
  __shared__ float gsh;

  for (int j = t; j < H_; j += 256) cat[j] = pooled[b * H_ + j];
  for (int j = t; j < P_; j += 256) cat[H_ + j] = hbuf[b * P_ + j];
  __syncthreads();

  if (t < P_) {
    float s = pb1[t];
    const float* w = pw1 + (long long)t * (H_ + P_);
    for (int k2 = 0; k2 < H_ + P_; ++k2) s += w[k2] * cat[k2];
    t1s[t] = 0.5f * s * (1.f + erff(s * 0.70710678118f));   // exact gelu
  }
  __syncthreads();
  if (t < P_) {
    float s = pb2[t];
    const float* w = pw2 + t * P_;
    for (int k2 = 0; k2 < P_; ++k2) s += w[k2] * t1s[k2];
    hm[t] = s;
  }
  __syncthreads();
  if (t < P_) {   // h += h_delta(h_mid)
    float s = hdb[t];
    const float* w = hdw + t * P_;
    for (int k2 = 0; k2 < P_; ++k2) s += w[k2] * hm[k2];
    hbuf[b * P_ + t] = cat[H_ + t] + s;
  }
  for (int j = t; j < 2 * H_; j += 256) {   // film
    float s = fb[j];
    const float* w = fw + (long long)j * P_;
    for (int k2 = 0; k2 < P_; ++k2) s += w[k2] * hm[k2];
    film_g[b * 2 * H_ + j] = s;
  }
  if (t < M_) {   // q, k, v
    float sq = mqb[t], sk = mkb[t], sv = mvb[t];
    const float* wq = mqw + t * P_;
    const float* wk = mkw + t * P_;
    const float* wv = mvw + t * P_;
    for (int k2 = 0; k2 < P_; ++k2) {
      sq += wq[k2] * hm[k2]; sk += wk[k2] * hm[k2]; sv += wv[k2] * hm[k2];
    }
    qv[t] = sq; kv[t] = sk; vv[t] = sv;
  }
  if (t == 0) {   // gate scalar + halt logit
    float sg = mgb[0], sh2 = hb[0];
    for (int k2 = 0; k2 < P_; ++k2) { sg += mgw[k2] * hm[k2]; sh2 += hw[k2] * hm[k2]; }
    gsh = sg;
    haltOut[b] = sh2;
  }
  __syncthreads();
  if (t < SLOTS_) {   // attention scores (read and write paths)
    const float* mrow = memw + ((long long)b * SLOTS_ + t) * M_;
    float s1 = 0.f, s2 = 0.f;
    for (int d = 0; d < M_; ++d) { s1 += mrow[d] * qv[d]; s2 += mrow[d] * kv[d]; }
    sc[t]  = s1 * 0.08838834764831845f;   // 1/sqrt(M)
    wsc[t] = s2 * 0.08838834764831845f;
  }
  __syncthreads();
  if (t == 0) {   // softmax over 64 slots
    float mx = -3.0e38f;
    for (int n = 0; n < SLOTS_; ++n) mx = fmaxf(mx, sc[n]);
    float sm = 0.f;
    for (int n = 0; n < SLOTS_; ++n) { sc[n] = __expf(sc[n] - mx); sm += sc[n]; }
    float inv = 1.f / sm;
    for (int n = 0; n < SLOTS_; ++n) sc[n] *= inv;
  }
  if (t == 32) {  // parallel softmax for write-attention on second wave
    float mx = -3.0e38f;
    for (int n = 0; n < SLOTS_; ++n) mx = fmaxf(mx, wsc[n]);
    float sm = 0.f;
    for (int n = 0; n < SLOTS_; ++n) { wsc[n] = __expf(wsc[n] - mx); sm += wsc[n]; }
    float inv = 1.f / sm;
    for (int n = 0; n < SLOTS_; ++n) wsc[n] *= inv;
  }
  __syncthreads();
  if (t < M_) {   // ctx_m = r_attn @ mem
    float s = 0.f;
    for (int n = 0; n < SLOTS_; ++n)
      s += sc[n] * memw[((long long)b * SLOTS_ + n) * M_ + t];
    ctxm[t] = s;
  }
  __syncthreads();
  for (int j = t; j < H_; j += 256) {   // ctx = memo(ctx_m)
    float s = mob[j];
    const float* w = mow + (long long)j * M_;
    for (int d = 0; d < M_; ++d) s += w[d] * ctxm[d];
    ctx_g[b * H_ + j] = s;
  }
  float gate = 1.f / (1.f + __expf(-gsh));
  for (int idx = t; idx < SLOTS_ * M_; idx += 256) {   // gated mem write
    int n = idx >> 7, d = idx & 127;
    long long a = ((long long)b * SLOTS_ + n) * M_ + d;
    float wn = gate * wsc[n];
    memw[a] = memw[a] * (1.f - wn) + wn * vv[d];
  }
}

// z += sqrt(H)*embed[inputs] + ctx
__global__ void zupd_k(float* __restrict__ z, const int* __restrict__ inp,
                       const float* __restrict__ embw, const float* __restrict__ ctx_g) {
  long long idx = (long long)blockIdx.x * blockDim.x + threadIdx.x;
  int hh = (int)(idx & (H_ - 1));
  long long bs = idx >> 9;          // b*S + s
  int b = (int)(bs >> 11);
  z[idx] += SQRT_H * embw[(long long)inp[bs] * H_ + hh] + ctx_g[b * H_ + hh];
}

// rms over sequence dim + transpose back + FiLM
__global__ __launch_bounds__(256) void rmsT_k(const float* __restrict__ tmp,
                                              const float* __restrict__ film_g,
                                              float* __restrict__ z) {
  __shared__ float sh[256];
  int bh = blockIdx.x, t = threadIdx.x;
  int b = bh >> 9, hh = bh & (H_ - 1);
  const float* row = tmp + (long long)bh * S_;
  float s = 0.f;
  for (int i = t; i < S_; i += 256) { float v = row[i]; s += v * v; }
  sh[t] = s; __syncthreads();
  for (int off = 128; off > 0; off >>= 1) {
    if (t < off) sh[t] += sh[t + off];
    __syncthreads();
  }
  float rinv = rsqrtf(sh[0] * (1.f / S_) + EPS_);
  float sc2 = 1.f + film_g[b * 2 * H_ + hh];
  float sf2 = film_g[b * 2 * H_ + H_ + hh];
  for (int i = t; i < S_; i += 256)
    z[(long long)b * S_ * H_ + (long long)i * H_ + hh] = row[i] * rinv * sc2 + sf2;
}

// rms over channel dim
__global__ __launch_bounds__(256) void rmsC_k(const float* __restrict__ tmp,
                                              float* __restrict__ z) {
  __shared__ float sh[256];
  long long r = blockIdx.x; int t = threadIdx.x;
  const float* row = tmp + r * H_;
  float v0 = row[t], v1 = row[t + 256];
  sh[t] = v0 * v0 + v1 * v1; __syncthreads();
  for (int off = 128; off > 0; off >>= 1) {
    if (t < off) sh[t] += sh[t + off];
    __syncthreads();
  }
  float rinv = rsqrtf(sh[0] * (1.f / H_) + EPS_);
  z[r * H_ + t]       = v0 * rinv;
  z[r * H_ + t + 256] = v1 * rinv;
}

// ---------------------------------------------------------------------------
extern "C" void kernel_launch(void* const* d_in, const int* in_sizes, int n_in,
                              void* d_out, int out_size, void* d_ws, size_t ws_size,
                              hipStream_t stream) {
  const int*   inp   = (const int*)  d_in[0];
  const float* zin   = (const float*)d_in[1];
  const float* hin   = (const float*)d_in[2];
  const float* memin = (const float*)d_in[3];
  const float* embw  = (const float*)d_in[4];
  const float* lmw   = (const float*)d_in[5];
  const float* pw1   = (const float*)d_in[6];
  const float* pb1   = (const float*)d_in[7];
  const float* pw2   = (const float*)d_in[8];
  const float* pb2   = (const float*)d_in[9];
  const float* hdw   = (const float*)d_in[10];
  const float* hdb   = (const float*)d_in[11];
  const float* fw    = (const float*)d_in[12];
  const float* fb    = (const float*)d_in[13];
  const float* hw    = (const float*)d_in[14];
  const float* hb    = (const float*)d_in[15];
  const float* mqw   = (const float*)d_in[16];
  const float* mqb   = (const float*)d_in[17];
  const float* mkw   = (const float*)d_in[18];
  const float* mkb   = (const float*)d_in[19];
  const float* mvw   = (const float*)d_in[20];
  const float* mvb   = (const float*)d_in[21];
  const float* mgw   = (const float*)d_in[22];
  const float* mgb   = (const float*)d_in[23];
  const float* mow   = (const float*)d_in[24];
  const float* mob   = (const float*)d_in[25];
  const float* wtgu  = (const float*)d_in[26];
  const float* wtd   = (const float*)d_in[27];
  const float* wmgu  = (const float*)d_in[28];
  const float* wmd   = (const float*)d_in[29];

  const long long ZN   = (long long)B_ * S_ * H_;     // 33,554,432
  const long long ACTN = (long long)B_ * S_ * IM_;    // 50,331,648 >= B*H*IT
  const long long N_TGU = 2LL * 2 * IT_ * S_;         // 23,068,672
  const long long N_TD  = 2LL * S_ * IT_;             // 11,534,336
  const long long N_MGU = 2LL * 2 * IM_ * H_;         //  1,572,864
  const long long N_MD  = 2LL * H_ * IM_;             //    786,432
  const long long N_LM  = (long long)V_ * H_;         //     65,536

  float* z      = (float*)d_ws;
  float* tmp    = z + ZN;
  float* pooled = tmp + ZN;
  float* hbuf   = pooled + (long long)B_ * H_;
  float* memw   = hbuf + (long long)B_ * P_;
  float* filmb  = memw + (long long)B_ * SLOTS_ * M_;
  float* ctxb   = filmb + (long long)B_ * 2 * H_;
  u16* act16    = (u16*)(ctxb + (long long)B_ * H_);
  u16* wtgu16   = act16 + ACTN;
  u16* wtd16    = wtgu16 + N_TGU;
  u16* wmgu16   = wtd16 + N_TD;
  u16* wmd16    = wmgu16 + N_MGU;
  u16* lm16     = wmd16 + N_MD;

  float* out = (float*)d_out;
  const long long LOG_N = (long long)4 * B_ * S_ * V_;
  float* haltBase = out + LOG_N;
  float* zOut = haltBase + 4 * B_;
  float* hOut = zOut + ZN;
  float* mOut = hOut + (long long)B_ * P_;

  (void)hipMemcpyAsync(z, zin, ZN * sizeof(float), hipMemcpyDeviceToDevice, stream);
  (void)hipMemcpyAsync(hbuf, hin, (size_t)B_ * P_ * sizeof(float), hipMemcpyDeviceToDevice, stream);
  (void)hipMemcpyAsync(memw, memin, (size_t)B_ * SLOTS_ * M_ * sizeof(float), hipMemcpyDeviceToDevice, stream);

  // One-time fp32 -> bf16 weight conversion (weights then L2-resident as bf16)
  cvt_k<<<(int)(N_TGU / 8 / 256), 256, 0, stream>>>(wtgu, wtgu16);
  cvt_k<<<(int)(N_TD  / 8 / 256), 256, 0, stream>>>(wtd,  wtd16);
  cvt_k<<<(int)(N_MGU / 8 / 256), 256, 0, stream>>>(wmgu, wmgu16);
  cvt_k<<<(int)(N_MD  / 8 / 256), 256, 0, stream>>>(wmd,  wmd16);
  cvt_k<<<(int)(N_LM  / 8 / 256), 256, 0, stream>>>(lmw,  lm16);

  for (int step = 0; step < 4; ++step) {
    pool_k<<<(B_ * H_) / 256, 256, 0, stream>>>(z, pooled);
    planner_k<<<B_, 256, 0, stream>>>(pooled, hbuf, memw, filmb, ctxb,
        haltBase + step * B_, pw1, pb1, pw2, pb2, hdw, hdb, fw, fb, hw, hb,
        mqw, mqb, mkw, mkb, mvw, mvb, mgw, mgb, mow, mob);
    zupd_k<<<(int)(ZN / 256), 256, 0, stream>>>(z, inp, embw, ctxb);

    for (int i = 0; i < L_; ++i) {
      // token mixer: act16 = bf16(silu(g)*u) over transposed z view
      gemm_wmma_k<0, false><<<dim3(IT_ / BLK_N, (B_ * H_) / BLK_M), 256, 0, stream>>>(
          z, nullptr, 0, wtgu16 + (long long)i * 2 * IT_ * S_,
          nullptr, act16,
          9, (long long)S_ * H_, 1, H_, S_, nullptr, IT_, IT_);
      gemm_wmma_k<1, true><<<dim3(S_ / BLK_N, (B_ * H_) / BLK_M), 256, 0, stream>>>(
          nullptr, act16, IT_, wtd16 + (long long)i * S_ * IT_,
          tmp, nullptr,
          40, 0, 0, 0, IT_, z, 0, S_);
      rmsT_k<<<B_ * H_, 256, 0, stream>>>(tmp, filmb, z);
      // channel mixer
      gemm_wmma_k<0, false><<<dim3(IM_ / BLK_N, (B_ * S_) / BLK_M), 256, 0, stream>>>(
          z, nullptr, 0, wmgu16 + (long long)i * 2 * IM_ * H_,
          nullptr, act16,
          40, 0, H_, 1, H_, nullptr, IM_, IM_);
      gemm_wmma_k<2, true><<<dim3(H_ / BLK_N, (B_ * S_) / BLK_M), 256, 0, stream>>>(
          nullptr, act16, IM_, wmd16 + (long long)i * H_ * IM_,
          tmp, nullptr,
          40, 0, 0, 0, IM_, z, 0, H_);
      rmsC_k<<<B_ * S_, 256, 0, stream>>>(tmp, z);
    }
    // lm_head -> logits_steps[step]
    gemm_wmma_k<3, false><<<dim3(V_ / BLK_N, (B_ * S_) / BLK_M), 256, 0, stream>>>(
        z, nullptr, 0, lm16,
        out + (long long)step * B_ * S_ * V_, nullptr,
        40, 0, H_, 1, H_, nullptr, 0, V_);
  }

  (void)hipMemcpyAsync(zOut, z, ZN * sizeof(float), hipMemcpyDeviceToDevice, stream);
  (void)hipMemcpyAsync(hOut, hbuf, (size_t)B_ * P_ * sizeof(float), hipMemcpyDeviceToDevice, stream);
  (void)hipMemcpyAsync(mOut, memw, (size_t)B_ * SLOTS_ * M_ * sizeof(float), hipMemcpyDeviceToDevice, stream);
}
